// MultiQueryAttention_32469952758061
// MI455X (gfx1250) — matrix operand order repaired and checked by reading, not compiled, not run on hardware
//
#include <hip/hip_runtime.h>
#include <hip/hip_bf16.h>
#include <math.h>

// ---------------------------------------------------------------------------
// MI455X (gfx1250) GQA attention block, bf16 WMMA everywhere.
//   q/k/v proj (WMMA) -> RoPE(+scale) -> flash attention (WMMA, softcap,
//   causal, async K staging) -> output proj (WMMA).
// Compute-bound (~155 GFLOP vs ~180MB of HBM traffic), so all six matmuls
// run on v_wmma_f32_16x16x32_bf16 with f32 accumulation. K tiles are staged
// LDS-side with global_load_async_to_lds_b128 (ASYNCcnt path).
// ---------------------------------------------------------------------------

typedef __attribute__((ext_vector_type(16))) __bf16 v16bf;
typedef __attribute__((ext_vector_type(8)))  float  v8f;

union Frag16 {
  v16bf bf;
  uint4 q[2];
};
static_assert(sizeof(v16bf) == 32, "v16bf must be 32 bytes");

__device__ __forceinline__ unsigned short f2bf(float x) {
  union { float f; unsigned u; } v; v.f = x;
  unsigned r = v.u + 0x7FFFu + ((v.u >> 16) & 1u);  // round-to-nearest-even
  return (unsigned short)(r >> 16);
}
__device__ __forceinline__ float bf2f(unsigned short u) {
  union { unsigned u; float f; } v; v.u = ((unsigned)u) << 16;
  return v.f;
}

// ---------------------------------------------------------------------------
// GEMM: C[MxN] = A[MxK](f32) * B[KxN](f32), computed in bf16 WMMA, f32 accum.
// Block 256 threads (8 waves), tile 128x64, k-step 32. Wave (wm,wn) in 4x2
// grid owns a 32x32 subtile = 2x2 WMMA tiles. f32 -> bf16 conversion happens
// while staging into LDS, so the raw f32 operands are read exactly once per
// tile pass (and A/B both fit in the 192MB L2 for re-reads).
// ---------------------------------------------------------------------------
#define BM 128
#define BN 64
#define BK 32

template <bool OUT_F32>
__global__ __launch_bounds__(256) void gemm_f32in_bf16_wmma(
    const float* __restrict__ A, const float* __restrict__ B,
    void* __restrict__ C, int M, int N, int K) {
  __shared__ __align__(16) unsigned short As[BM * BK];  // [m][k]      8 KB
  __shared__ __align__(16) unsigned short Bt[BN * BK];  // [n][k] (T)  4 KB

  const int tid  = threadIdx.x;
  const int lane = tid & 31;
  const int hi   = lane >> 4;
  const int lm   = lane & 15;
  const int w    = tid >> 5;
  const int wm   = w >> 1;   // 0..3
  const int wn   = w & 1;    // 0..1
  const int m0   = blockIdx.y * BM;
  const int n0   = blockIdx.x * BN;

  const v8f zero8 = {0.f, 0.f, 0.f, 0.f, 0.f, 0.f, 0.f, 0.f};
  v8f acc[2][2];
#pragma unroll
  for (int mi = 0; mi < 2; ++mi)
#pragma unroll
    for (int ni = 0; ni < 2; ++ni) acc[mi][ni] = zero8;

  const int arow  = tid >> 1;          // 0..127
  const int akseg = (tid & 1) * 16;    // 0 or 16
  const int brow  = tid >> 3;          // 0..31
  const int bnseg = (tid & 7) * 8;     // 0..56

  for (int kk = 0; kk < K; kk += BK) {
    // ---- stage A tile (convert f32 -> bf16) ----
    const float* ag = A + (size_t)(m0 + arow) * K + kk + akseg;
    union { unsigned short s[16]; uint4 q[2]; } ta;
#pragma unroll
    for (int i = 0; i < 4; ++i) {
      float4 v = *(const float4*)(ag + i * 4);
      ta.s[i * 4 + 0] = f2bf(v.x);
      ta.s[i * 4 + 1] = f2bf(v.y);
      ta.s[i * 4 + 2] = f2bf(v.z);
      ta.s[i * 4 + 3] = f2bf(v.w);
    }
    *(uint4*)(As + arow * BK + akseg)     = ta.q[0];
    *(uint4*)(As + arow * BK + akseg + 8) = ta.q[1];

    // ---- stage B tile transposed (convert f32 -> bf16) ----
    const float* bg = B + (size_t)(kk + brow) * N + n0 + bnseg;
    float4 b0 = *(const float4*)(bg);
    float4 b1 = *(const float4*)(bg + 4);
    unsigned short bs[8] = {f2bf(b0.x), f2bf(b0.y), f2bf(b0.z), f2bf(b0.w),
                            f2bf(b1.x), f2bf(b1.y), f2bf(b1.z), f2bf(b1.w)};
#pragma unroll
    for (int i = 0; i < 8; ++i) Bt[(bnseg + i) * BK + brow] = bs[i];

    if (kk + BK < K) {  // global_prefetch_b8 of next k-slab
      __builtin_prefetch(ag + BK, 0, 1);
      __builtin_prefetch(bg + (size_t)BK * N, 0, 1);
    }
    __syncthreads();

    // ---- fragments + WMMA ----
    Frag16 af[2], bfr[2];
#pragma unroll
    for (int mi = 0; mi < 2; ++mi) {
      const unsigned short* p = As + (wm * 32 + mi * 16 + lm) * BK;
      af[mi].q[0] = *(const uint4*)(p + hi * 8);
      af[mi].q[1] = *(const uint4*)(p + 16 + hi * 8);
    }
#pragma unroll
    for (int ni = 0; ni < 2; ++ni) {
      const unsigned short* p = Bt + (wn * 32 + ni * 16 + lm) * BK + hi * 8;
      bfr[ni].q[0] = *(const uint4*)(p);
      bfr[ni].q[1] = *(const uint4*)(p + 16);
    }
#pragma unroll
    for (int mi = 0; mi < 2; ++mi)
#pragma unroll
      for (int ni = 0; ni < 2; ++ni)
        acc[mi][ni] = __builtin_amdgcn_wmma_f32_16x16x32_bf16(
            false, af[mi].bf, false, bfr[ni].bf, (short)0, acc[mi][ni],
            false, false);
    __syncthreads();
  }

  // ---- epilogue: C layout m = r + hi*8, n = lm ----
#pragma unroll
  for (int mi = 0; mi < 2; ++mi)
#pragma unroll
    for (int ni = 0; ni < 2; ++ni)
#pragma unroll
      for (int r = 0; r < 8; ++r) {
        const int m = m0 + wm * 32 + mi * 16 + hi * 8 + r;
        const int n = n0 + wn * 32 + ni * 16 + lm;
        const float val = acc[mi][ni][r];
        if (OUT_F32)
          ((float*)C)[(size_t)m * N + n] = val;
        else
          ((unsigned short*)C)[(size_t)m * N + n] = f2bf(val);
      }
}

// ---------------------------------------------------------------------------
// In-place NeoX RoPE on bf16 head tensor (B, L, nh, 128); scale folded in
// (Q gets 1/sqrt(128), K gets 1.0).
// ---------------------------------------------------------------------------
__global__ __launch_bounds__(256) void rope_scale_kernel(
    unsigned short* __restrict__ buf, int nh, int total, float scale) {
  const int idx = blockIdx.x * blockDim.x + threadIdx.x;
  if (idx >= total) return;
  const int i = idx & 63;
  int t = idx >> 6;
  const int head = t % nh; t /= nh;
  const int pos = t % 2048;
  const int b = t / 2048;
  // inv_freq[i] = 10000^(-i/64) = exp(-i * ln(10000)/64)
  const float freq = __expf(-0.14391156f * (float)i);
  float s, c;
  __sincosf((float)pos * freq, &s, &c);
  const size_t base = (((size_t)b * 2048 + pos) * nh + head) * 128 + i;
  const float x1 = bf2f(buf[base]);
  const float x2 = bf2f(buf[base + 64]);
  buf[base]      = f2bf((x1 * c - x2 * s) * scale);
  buf[base + 64] = f2bf((x1 * s + x2 * c) * scale);
}

// ---------------------------------------------------------------------------
// Flash attention: block = 128 threads (4 waves x 16 query rows), key blocks
// of 32 staged in LDS. K tile uses the gfx1250 async global->LDS path
// (ASYNCcnt); V tile is transposed manually while the async copy runs.
// Softcap 50*tanh(s/50), analytic causal mask, online softmax. QK^T and PV
// both on bf16 WMMA.
// ---------------------------------------------------------------------------
#define L_SEQ  2048
#define DMODEL 2048
#define KVDIM  512

__global__ __launch_bounds__(128) void attn_fwd_kernel(
    const unsigned short* __restrict__ Qp,
    const unsigned short* __restrict__ Kp,
    const unsigned short* __restrict__ Vp,
    float* __restrict__ Op) {
  __shared__ __align__(16) unsigned short Ks[32 * 128];   // K block [k][d] 8 KB
  __shared__ __align__(16) unsigned short Vt[128 * 32];   // V blk^T [d][k] 8 KB
  __shared__ __align__(16) unsigned short Ps[4 * 16 * 32];// P per wave     4 KB

  const int tid  = threadIdx.x;
  const int lane = tid & 31;
  const int hi   = lane >> 4;
  const int lm   = lane & 15;
  const int w    = tid >> 5;
  const int b    = blockIdx.z;
  const int h    = blockIdx.y;
  const int kvh  = h >> 2;                // REP = 4
  const int qblk = blockIdx.x * 64;
  const int q0   = qblk + w * 16;

  // ---- load Q A-fragments for the wave's 16 rows (already rope'd+scaled) ---
  Frag16 qf[4];
  {
    const unsigned short* qrow =
        Qp + (((size_t)b * L_SEQ + q0 + lm) * DMODEL + h * 128);
#pragma unroll
    for (int s = 0; s < 4; ++s) {
      qf[s].q[0] = *(const uint4*)(qrow + s * 32 + hi * 8);
      qf[s].q[1] = *(const uint4*)(qrow + s * 32 + 16 + hi * 8);
    }
  }

  const v8f zero8 = {0.f, 0.f, 0.f, 0.f, 0.f, 0.f, 0.f, 0.f};
  v8f o[8];
#pragma unroll
  for (int j = 0; j < 8; ++j) o[j] = zero8;
  float rm[8], rs[8];
#pragma unroll
  for (int r = 0; r < 8; ++r) { rm[r] = -1e30f; rs[r] = 0.f; }

  const int cprow = tid >> 2;           // 0..31 (copy row of the key block)
  const int cpseg = (tid & 3) * 32;     // 0/32/64/96 (dim segment)
  // Per-lane LDS byte offset for the async K copy. Generic LDS pointers carry
  // the LDS offset in their low 32 bits.
  const unsigned ks_lds =
      (unsigned)(uintptr_t)(Ks + cprow * 128 + cpseg);

  const int nkb = (qblk >> 5) + 2;  // key blocks up to the block diagonal
  for (int kb = 0; kb < nkb; ++kb) {
    const int k0 = kb * 32;
    __syncthreads();  // previous iteration's LDS reads complete

    // ---- K tile: async global -> LDS copy (4 x b128 per lane, ASYNCcnt) ----
    {
      const unsigned short* ksrc =
          Kp + (((size_t)b * L_SEQ + k0 + cprow) * KVDIM + kvh * 128 + cpseg);
#pragma unroll
      for (int i = 0; i < 4; ++i) {
        asm volatile("global_load_async_to_lds_b128 %0, %1, off"
                     :
                     : "v"(ks_lds + (unsigned)(i * 16)), "v"(ksrc + i * 8)
                     : "memory");
      }
    }
    // ---- V tile: load + transpose into Vt while the async copy runs ----
    {
      const unsigned short* vsrc =
          Vp + (((size_t)b * L_SEQ + k0 + cprow) * KVDIM + kvh * 128 + cpseg);
      union { unsigned short s[32]; uint4 q[4]; } vl;
#pragma unroll
      for (int i = 0; i < 4; ++i) vl.q[i] = *(const uint4*)(vsrc + i * 8);
#pragma unroll
      for (int i = 0; i < 32; ++i) Vt[(cpseg + i) * 32 + cprow] = vl.s[i];
    }
    asm volatile("s_wait_asynccnt 0x0" ::: "memory");
    __syncthreads();

    if (k0 > q0 + 15) continue;  // wave fully above diagonal: barriers done

    // ---- scores S = Q * K^T (two 16x16 n-tiles, 4 k-steps of 32) ----
    v8f sc[2];
    sc[0] = zero8; sc[1] = zero8;
#pragma unroll
    for (int nt = 0; nt < 2; ++nt)
#pragma unroll
      for (int s = 0; s < 4; ++s) {
        Frag16 bk;
        const unsigned short* kp = Ks + (nt * 16 + lm) * 128 + s * 32 + hi * 8;
        bk.q[0] = *(const uint4*)kp;
        bk.q[1] = *(const uint4*)(kp + 16);
        sc[nt] = __builtin_amdgcn_wmma_f32_16x16x32_bf16(
            false, qf[s].bf, false, bk.bf, (short)0, sc[nt], false, false);
      }

    // ---- softcap + causal mask + online softmax ----
    float pv0[8], pv1[8];
#pragma unroll
    for (int r = 0; r < 8; ++r) {
      const int row = q0 + hi * 8 + r;
      float s0 = 50.0f * tanhf(sc[0][r] * 0.02f);
      float s1 = 50.0f * tanhf(sc[1][r] * 0.02f);
      if (k0 + lm > row)      s0 = -1e30f;
      if (k0 + 16 + lm > row) s1 = -1e30f;
      float bm = fmaxf(s0, s1);
#pragma unroll
      for (int m = 1; m < 16; m <<= 1) bm = fmaxf(bm, __shfl_xor(bm, m, 32));
      const float mnew  = fmaxf(rm[r], bm);
      const float alpha = __expf(rm[r] - mnew);
      rm[r] = mnew;
      const float p0 = __expf(s0 - mnew);
      const float p1 = __expf(s1 - mnew);
      pv0[r] = p0; pv1[r] = p1;
      float ps = p0 + p1;
#pragma unroll
      for (int m = 1; m < 16; m <<= 1) ps += __shfl_xor(ps, m, 32);
      rs[r] = rs[r] * alpha + ps;
#pragma unroll
      for (int j = 0; j < 8; ++j) o[j][r] *= alpha;
    }

    // ---- P: C-layout -> bf16 A-layout via wave-private LDS ----
    unsigned short* pws = Ps + w * 512;
#pragma unroll
    for (int r = 0; r < 8; ++r) {
      pws[(hi * 8 + r) * 32 + lm]      = f2bf(pv0[r]);
      pws[(hi * 8 + r) * 32 + 16 + lm] = f2bf(pv1[r]);
    }
    Frag16 pa;
    const unsigned short* prow = pws + lm * 32;
    pa.q[0] = *(const uint4*)(prow + hi * 8);
    pa.q[1] = *(const uint4*)(prow + 16 + hi * 8);

    // ---- O += P * V (8 n-tiles over DK=128, K=32 single step) ----
#pragma unroll
    for (int j = 0; j < 8; ++j) {
      Frag16 bv;
      const unsigned short* vp = Vt + (j * 16 + lm) * 32 + hi * 8;
      bv.q[0] = *(const uint4*)vp;
      bv.q[1] = *(const uint4*)(vp + 16);
      o[j] = __builtin_amdgcn_wmma_f32_16x16x32_bf16(
          false, pa.bf, false, bv.bf, (short)0, o[j], false, false);
    }
  }

  // ---- normalize and write f32 (B, L, H, DK) for the output projection ----
#pragma unroll
  for (int r = 0; r < 8; ++r) {
    const float inv = 1.0f / rs[r];
    const int row = q0 + hi * 8 + r;
    float* orow = Op + (((size_t)b * L_SEQ + row) * DMODEL + h * 128);
#pragma unroll
    for (int j = 0; j < 8; ++j) orow[j * 16 + lm] = o[j][r] * inv;
  }
}

// ---------------------------------------------------------------------------
// Launch. Workspace layout (ws >= ~59 MB):
//   Qp bf16 [B*L*2048]  16.78 MB
//   Kp bf16 [B*L*512]    4.19 MB
//   Vp bf16 [B*L*512]    4.19 MB
//   Ao f32  [B*L*2048]  33.55 MB
// ---------------------------------------------------------------------------
extern "C" void kernel_launch(void* const* d_in, const int* in_sizes, int n_in,
                              void* d_out, int out_size, void* d_ws,
                              size_t ws_size, hipStream_t stream) {
  (void)in_sizes; (void)n_in; (void)out_size; (void)ws_size;
  const float* query = (const float*)d_in[0];
  const float* key   = (const float*)d_in[1];
  const float* value = (const float*)d_in[2];
  // d_in[3] is the causal tril mask; implemented analytically in-kernel.
  const float* w_q = (const float*)d_in[4];
  const float* w_k = (const float*)d_in[5];
  const float* w_v = (const float*)d_in[6];
  const float* w_o = (const float*)d_in[7];

  const int Bb = 2, L = 2048, D = 2048, KVD = 512;
  const int M = Bb * L;  // 4096

  unsigned short* Qp = (unsigned short*)d_ws;
  unsigned short* Kp = Qp + (size_t)M * D;
  unsigned short* Vp = Kp + (size_t)M * KVD;
  float*          Ao = (float*)(Vp + (size_t)M * KVD);

  // Projections (bf16 out).
  gemm_f32in_bf16_wmma<false>
      <<<dim3(D / BN, M / BM), dim3(256), 0, stream>>>(query, w_q, Qp, M, D, D);
  gemm_f32in_bf16_wmma<false>
      <<<dim3(KVD / BN, M / BM), dim3(256), 0, stream>>>(key, w_k, Kp, M, KVD, D);
  gemm_f32in_bf16_wmma<false>
      <<<dim3(KVD / BN, M / BM), dim3(256), 0, stream>>>(value, w_v, Vp, M, KVD, D);

  // RoPE (+ 1/sqrt(DK) folded into Q).
  const int totQ = Bb * L * 16 * 64;
  rope_scale_kernel<<<(totQ + 255) / 256, 256, 0, stream>>>(
      Qp, 16, totQ, 0.08838834764831845f);
  const int totK = Bb * L * 4 * 64;
  rope_scale_kernel<<<(totK + 255) / 256, 256, 0, stream>>>(Kp, 4, totK, 1.0f);

  // Flash attention.
  attn_fwd_kernel<<<dim3(L / 64, 16, Bb), dim3(128), 0, stream>>>(Qp, Kp, Vp, Ao);

  // Output projection (f32 out).
  gemm_f32in_bf16_wmma<true>
      <<<dim3(D / BN, M / BM), dim3(256), 0, stream>>>(Ao, w_o, (float*)d_out, M, D, D);
}
